// Encoder_6236292513890
// MI455X (gfx1250) — matrix operand order repaired and checked by reading, not compiled, not run on hardware
//
#include <hip/hip_runtime.h>
#include <hip/hip_bf16.h>

typedef __attribute__((ext_vector_type(16))) __bf16 v16bf;
typedef __attribute__((ext_vector_type(8)))  __bf16 v8bf;
typedef __attribute__((ext_vector_type(8)))  float  v8f;

#define GN_NODES 10000
#define GN_EDGES 320000
#define G_INF    512
#define G_H1     256
#define G_H2     64

// ---------------------------------------------------------------------------
// f32 -> bf16 elementwise cast
// ---------------------------------------------------------------------------
__global__ void k_cast_bf16(const float* __restrict__ src, __bf16* __restrict__ dst, int n) {
    int i = blockIdx.x * blockDim.x + threadIdx.x;
    if (i < n) dst[i] = (__bf16)src[i];
}

// ---------------------------------------------------------------------------
// Pack row-major W[K][N] (f32) into per-lane WMMA B-fragment order (bf16).
// Fragment layout (wave32, 16-bit B 32x16): lane holds column n = lane&15,
// halves i=0..15 <-> K = (i>>3)*16 + (lane>>4)*8 + (i&7) within a 32-deep K-tile.
// Output index o = (((ntile*Kt + ktile)*32 + lane)*16 + i)
// ---------------------------------------------------------------------------
__global__ void k_pack_b(const float* __restrict__ W, __bf16* __restrict__ Wp, int K, int N) {
    int o = blockIdx.x * blockDim.x + threadIdx.x;
    if (o >= K * N) return;
    int Kt    = K >> 5;
    int i     = o & 15;
    int lane  = (o >> 4) & 31;
    int t     = o >> 9;
    int ktile = t % Kt;
    int ntile = t / Kt;
    int k = ktile * 32 + ((i >> 3) << 4) + ((lane >> 4) << 3) + (i & 7);
    int n = ntile * 16 + (lane & 15);
    Wp[o] = (__bf16)W[k * N + n];
}

// ---------------------------------------------------------------------------
// C[M][N] = A[M][K] @ Bpacked, one 16x16 tile per wave, K stepped by 32.
// K,N are compile-time so the K-loop has a constant trip count and all
// fragment addresses fold to immediate offsets. The last iteration is peeled
// so the steady-state body is branch-free: shuffle current, load next, WMMA.
// blockDim = 128 (4 waves): wave w -> ntile = blockIdx.y*4 + w, mtile = blockIdx.x
// ---------------------------------------------------------------------------
template <int K, int N>
__global__ void k_gemm_bf16(const __bf16* __restrict__ A, const __bf16* __restrict__ Bp,
                            float* __restrict__ C) {
    constexpr int Kt = K >> 5;
    constexpr int Nt = N >> 4;
    const int lane  = threadIdx.x & 31;
    const int wave  = threadIdx.x >> 5;
    const int ntile = blockIdx.y * 4 + wave;
    if (ntile >= Nt) return;                 // wave-uniform: EXEC stays all-1s for WMMA
    const int mtile = blockIdx.x;
    const int arow  = mtile * 16 + (lane & 15);
    const int kb8   = (lane >> 4) << 3;      // A fragment: half-wave K split

    const __bf16* ap = A + (size_t)arow * K + kb8;
    const __bf16* bp = Bp + ((size_t)ntile * Kt * 32 + lane) * 16;

    v8bf  lo = *(const v8bf*)ap;             // K offsets [kb8, kb8+8)
    v8bf  hi = *(const v8bf*)(ap + 16);      // K offsets [kb8+16, kb8+24)
    v16bf bv = *(const v16bf*)bp;

    v8f acc = {};
#pragma unroll 4
    for (int kt = 0; kt < Kt - 1; ++kt) {    // steady state: branch-free body
        v16bf av = __builtin_shufflevector(lo, hi, 0,1,2,3,4,5,6,7,8,9,10,11,12,13,14,15);
        v16bf bc = bv;
        ap += 32;                            // +32 K (bf16)
        bp += 512;                           // next packed 32x16 B tile (1024 B)
        lo = *(const v8bf*)ap;
        hi = *(const v8bf*)(ap + 16);
        bv = *(const v16bf*)bp;
        acc = __builtin_amdgcn_wmma_f32_16x16x32_bf16(false, av, false, bc,
                                                      (short)0, acc, false, false);
    }
    {                                        // peeled final K-tile
        v16bf av = __builtin_shufflevector(lo, hi, 0,1,2,3,4,5,6,7,8,9,10,11,12,13,14,15);
        acc = __builtin_amdgcn_wmma_f32_16x16x32_bf16(false, av, false, bv,
                                                      (short)0, acc, false, false);
    }

    const int crow0 = mtile * 16 + ((lane >> 4) << 3);
    const int ccol  = ntile * 16 + (lane & 15);
#pragma unroll
    for (int r = 0; r < 8; ++r)
        C[(size_t)(crow0 + r) * N + ccol] = acc[r];
}

// ---------------------------------------------------------------------------
// agg[dst[e]][f] += w[e] * sup[src[e]][f]  -- one block per edge, thread per feat
// ---------------------------------------------------------------------------
__global__ void k_scatter(const float* __restrict__ sup, const int* __restrict__ src,
                          const int* __restrict__ dst, const float* __restrict__ w,
                          float* __restrict__ agg, int F) {
    int e = blockIdx.x;
    int f = threadIdx.x;
    float v = w[e] * sup[(size_t)src[e] * F + f];
    atomicAdd(&agg[(size_t)dst[e] * F + f], v);
}

// ---------------------------------------------------------------------------
// h = relu(agg + b); write bf16 always, f32 optionally (layer-2 z -> d_out)
// F must be a power of two.
// ---------------------------------------------------------------------------
__global__ void k_bias_relu(const float* __restrict__ agg, const float* __restrict__ b,
                            float* __restrict__ outf, __bf16* __restrict__ outb,
                            int total, int F) {
    int i = blockIdx.x * blockDim.x + threadIdx.x;
    if (i >= total) return;
    float v = agg[i] + b[i & (F - 1)];
    v = v > 0.0f ? v : 0.0f;
    if (outf) outf[i] = v;
    outb[i] = (__bf16)v;
}

// ---------------------------------------------------------------------------
// recon = sigmoid(z @ z.T); z is [N_NODES][64] bf16 row-major.
// B column n == z row n -> contiguous fragment loads for both operands.
// Output is 400 MB write-once (> 192 MB L2): non-temporal stores keep the
// streaming result from evicting the fully-L2-resident z operand.
// Sigmoid uses v_exp_f32 + v_rcp_f32 (1 ulp) instead of the ~10-op IEEE
// division expansion: 1e8 elements must stay under the HBM store roof.
// ---------------------------------------------------------------------------
__global__ void k_gram_sigmoid(const __bf16* __restrict__ zb, float* __restrict__ out) {
    const int lane  = threadIdx.x & 31;
    const int wave  = threadIdx.x >> 5;
    const int NT    = GN_NODES / 16;         // 625
    const int ntile = blockIdx.y * 4 + wave;
    if (ntile >= NT) return;                 // wave-uniform
    const int mtile = blockIdx.x;
    const int arow  = mtile * 16 + (lane & 15);
    const int brow  = ntile * 16 + (lane & 15);
    const int kb8   = (lane >> 4) << 3;
    const int kb16  = (lane >> 4) << 4;

    const __bf16* ap = zb + (size_t)arow * G_H2 + kb16;
    const __bf16* bp = zb + (size_t)brow * G_H2 + kb16;

    v8f acc = {};
#pragma unroll
    for (int kt = 0; kt < 2; ++kt) {         // K = 64 = 2 x 32
        v8bf lo = *(const v8bf*)(ap + kt * 32);
        v8bf hi = *(const v8bf*)(ap + kt * 32 + 8);
        v16bf av = __builtin_shufflevector(lo, hi, 0,1,2,3,4,5,6,7,8,9,10,11,12,13,14,15);
        v16bf bv = *(const v16bf*)(bp + kt * 32);
        acc = __builtin_amdgcn_wmma_f32_16x16x32_bf16(false, av, false, bv,
                                                      (short)0, acc, false, false);
    }
    const int row0 = mtile * 16 + ((lane >> 4) << 3);
    const int col  = ntile * 16 + (lane & 15);
#pragma unroll
    for (int r = 0; r < 8; ++r) {
        float s = __builtin_amdgcn_rcpf(1.0f + __expf(-acc[r]));  // v_rcp_f32
        __builtin_nontemporal_store(s, &out[(size_t)(row0 + r) * GN_NODES + col]);
    }
}

// ---------------------------------------------------------------------------
extern "C" void kernel_launch(void* const* d_in, const int* in_sizes, int n_in,
                              void* d_out, int out_size, void* d_ws, size_t ws_size,
                              hipStream_t stream) {
    const float* x    = (const float*)d_in[0];
    const float* W1   = (const float*)d_in[1];
    const float* b1   = (const float*)d_in[2];
    const float* W2   = (const float*)d_in[3];
    const float* b2   = (const float*)d_in[4];
    const float* ew   = (const float*)d_in[5];
    const int*   esrc = (const int*)d_in[6];
    const int*   edst = (const int*)d_in[7];
    float* out = (float*)d_out;

    char* ws = (char*)d_ws;
    size_t off = 0;
    auto carve = [&](size_t bytes) -> void* {
        void* p = ws + off;
        off += (bytes + 255) & ~(size_t)255;
        return p;
    };
    __bf16* xb   = (__bf16*)carve((size_t)GN_NODES * G_INF * 2);  // 10.24 MB
    __bf16* W1p  = (__bf16*)carve((size_t)G_INF * G_H1 * 2);
    float*  sup1 = (float*) carve((size_t)GN_NODES * G_H1 * 4);
    float*  agg1 = (float*) carve((size_t)GN_NODES * G_H1 * 4);
    __bf16* h1b  = (__bf16*)carve((size_t)GN_NODES * G_H1 * 2);
    __bf16* W2p  = (__bf16*)carve((size_t)G_H1 * G_H2 * 2);
    float*  sup2 = (float*) carve((size_t)GN_NODES * G_H2 * 4);
    float*  agg2 = (float*) carve((size_t)GN_NODES * G_H2 * 4);
    __bf16* zb   = (__bf16*)carve((size_t)GN_NODES * G_H2 * 2);   // total ~43 MB

    hipMemsetAsync(agg1, 0, (size_t)GN_NODES * G_H1 * 4, stream);
    hipMemsetAsync(agg2, 0, (size_t)GN_NODES * G_H2 * 4, stream);

    // Layer 1
    k_cast_bf16<<<(GN_NODES * G_INF + 255) / 256, 256, 0, stream>>>(x, xb, GN_NODES * G_INF);
    k_pack_b  <<<(G_INF * G_H1 + 255) / 256, 256, 0, stream>>>(W1, W1p, G_INF, G_H1);
    k_gemm_bf16<G_INF, G_H1><<<dim3(GN_NODES / 16, (G_H1 / 16 + 3) / 4), 128, 0, stream>>>(
        xb, W1p, sup1);
    k_scatter<<<GN_EDGES, G_H1, 0, stream>>>(sup1, esrc, edst, ew, agg1, G_H1);
    k_bias_relu<<<(GN_NODES * G_H1 + 255) / 256, 256, 0, stream>>>(
        agg1, b1, nullptr, h1b, GN_NODES * G_H1, G_H1);

    // Layer 2 (z = relu(agg2 + b2) -> f32 into d_out[0 .. 640000), bf16 copy for decoder)
    k_pack_b<<<(G_H1 * G_H2 + 255) / 256, 256, 0, stream>>>(W2, W2p, G_H1, G_H2);
    k_gemm_bf16<G_H1, G_H2><<<dim3(GN_NODES / 16, (G_H2 / 16 + 3) / 4), 128, 0, stream>>>(
        h1b, W2p, sup2);
    k_scatter<<<GN_EDGES, G_H2, 0, stream>>>(sup2, esrc, edst, ew, agg2, G_H2);
    k_bias_relu<<<(GN_NODES * G_H2 + 255) / 256, 256, 0, stream>>>(
        agg2, b2, out, zb, GN_NODES * G_H2, G_H2);

    // Decoder: recon = sigmoid(z z^T) -> d_out[640000 ..)
    k_gram_sigmoid<<<dim3(GN_NODES / 16, (GN_NODES / 16 + 3) / 4), 128, 0, stream>>>(
        zb, out + (size_t)GN_NODES * G_H2);
}
